// EffiGlobalPointer_6554120094166
// MI455X (gfx1250) — matrix operand order repaired and checked by reading, not compiled, not run on hardware
//
#include <hip/hip_runtime.h>
#include <hip/hip_bf16.h>

// ---- problem constants ----
#define BB    8
#define NN    1024
#define HH    768
#define INNER 64
#define ENT   9
#define NEGV  1000000000000.0f
#define WPAD  160           // 128 (w1) + 18 (w2) padded to 10*16
#define KT1   (HH / 32)     // 24 k-tiles for projection

typedef __attribute__((ext_vector_type(16))) __bf16 v16bf;  // 32 B
typedef __attribute__((ext_vector_type(8)))  __bf16 v8bf;   // 16 B
typedef __attribute__((ext_vector_type(4)))  __bf16 v4bf;   // 8 B
typedef __attribute__((ext_vector_type(8)))  float  v8f;
typedef __attribute__((ext_vector_type(4)))  float  v4f;

// ---------------------------------------------------------------------------
// Kernel 0a: pack [w1 | w2 | zeros] TRANSPOSED -> WcT (160 cols x 768 k), bf16.
// K is contiguous per column so B fragments are single 32B vector loads.
// ---------------------------------------------------------------------------
__global__ void k_prep_w(const float* __restrict__ w1, const float* __restrict__ w2,
                         __bf16* __restrict__ wcT) {
    int tid = blockIdx.x * blockDim.x + threadIdx.x;
    if (tid >= WPAD * HH) return;
    int c = tid / HH, k = tid % HH;
    float v = 0.0f;
    if (c < 128)              v = w1[k * 128 + c];
    else if (c < 128 + 2*ENT) v = w2[k * (2*ENT) + (c - 128)];
    wcT[tid] = (__bf16)v;
}

// ---------------------------------------------------------------------------
// Kernel 0b: convert lhs (8192 x 768 f32) -> bf16, 4 elements/thread.
// ---------------------------------------------------------------------------
__global__ void k_prep_x(const float* __restrict__ lhs, __bf16* __restrict__ lhsb) {
    int tid = blockIdx.x * blockDim.x + threadIdx.x;      // (8192*768)/4
    if (tid >= (BB * NN * HH) / 4) return;
    v4f x = *(const v4f*)(lhs + (size_t)tid * 4);
    v4bf y;
    y[0] = (__bf16)x[0]; y[1] = (__bf16)x[1];
    y[2] = (__bf16)x[2]; y[3] = (__bf16)x[3];
    *(v4bf*)(lhsb + (size_t)tid * 4) = y;
}

// ---------------------------------------------------------------------------
// Kernel 1: projection GEMM  Y(8192 x 160) = lhs @ [w1|w2] + bias.
// One wave per 16-row M tile, 10 N-tile accumulators in VGPRs.
// A fragment: two contiguous v8bf loads; B fragment: one v16bf load.
// ---------------------------------------------------------------------------
__global__ void k_proj(const __bf16* __restrict__ lhsb, const __bf16* __restrict__ wcT,
                       const float* __restrict__ b1, const float* __restrict__ b2,
                       float* __restrict__ Y) {
    int wave = (blockIdx.x * blockDim.x + threadIdx.x) >> 5;   // 0..511
    int lane = threadIdx.x & 31;
    int lm = lane & 15, lh = lane >> 4;
    int m = wave * 16 + lm;                                    // A-matrix row

    v8f acc[10];
#pragma unroll
    for (int t = 0; t < 10; ++t) acc[t] = (v8f){0,0,0,0,0,0,0,0};

    const __bf16* abase = lhsb + (size_t)m * HH + lh * 16;
    for (int kt = 0; kt < KT1; ++kt) {
        int kbase = kt * 32;
        // A fragment 16x32 bf16: lanes 0-15 hold K 0-15 (lh selects +16)
        v8bf alo = *(const v8bf*)(abase + kbase);
        v8bf ahi = *(const v8bf*)(abase + kbase + 8);
        v16bf a = __builtin_shufflevector(alo, ahi,
                     0,1,2,3,4,5,6,7,8,9,10,11,12,13,14,15);
        if (kt + 1 < KT1) __builtin_prefetch(abase + kbase + 32, 0, 1);

#pragma unroll
        for (int t = 0; t < 10; ++t) {
            int col = t * 16 + lm;                             // B-matrix column
            v16bf bm = *(const v16bf*)(wcT + (size_t)col * HH + kbase + lh * 16);
            acc[t] = __builtin_amdgcn_wmma_f32_16x16x32_bf16(
                false, a, false, bm, (short)0, acc[t], false, false);
        }
    }

    // Epilogue: add bias, store Y tile. C layout: row = r + 8*lh, col = lm.
#pragma unroll
    for (int t = 0; t < 10; ++t) {
        int col = t * 16 + lm;
        float bias = (col < 128) ? b1[col]
                   : ((col < 128 + 2*ENT) ? b2[col - 128] : 0.0f);
#pragma unroll
        for (int r = 0; r < 8; ++r) {
            int row = wave * 16 + r + 8 * lh;
            Y[(size_t)row * WPAD + col] = acc[t][r] + bias;
        }
    }
}

// ---------------------------------------------------------------------------
// Kernel 2: RoPE + even/odd split -> bf16 qw, kw  (each 8192 x 64).
// The 4 floats a thread needs (q/k pair group) are one aligned float4.
// ---------------------------------------------------------------------------
__global__ void k_rope(const float* __restrict__ Y,
                       __bf16* __restrict__ qwr, __bf16* __restrict__ kwr) {
    int tid = blockIdx.x * blockDim.x + threadIdx.x;           // 8192*64
    if (tid >= BB * NN * INNER) return;
    int row = tid >> 6, c = tid & 63;
    int pos = row & (NN - 1);
    int i = c >> 1;
    // inv_freq = 10000^(-2i/64) = exp(-i * ln(10000)/32)
    float inv = __expf(-0.28782313662425572f * (float)i);
    float ang = (float)pos * inv;
    float s = __sinf(ang), co = __cosf(ang);
    // group j = c>>1 covers Y cols [4j..4j+3] = {q[2j], k[2j], q[2j+1], k[2j+1]}
    v4f g = *(const v4f*)(Y + (size_t)row * WPAD + 4 * i);
    float qc, qp, kc, kp, sgn;
    if (c & 1) { qc = g[2]; qp = g[0]; kc = g[3]; kp = g[1]; sgn =  1.0f; }
    else       { qc = g[0]; qp = g[2]; kc = g[1]; kp = g[3]; sgn = -1.0f; }
    qwr[tid] = (__bf16)(qc * co + sgn * qp * s);
    kwr[tid] = (__bf16)(kc * co + sgn * kp * s);
}

// ---------------------------------------------------------------------------
// Kernel 3: extract entity biases (layout (b, e, pos)), divided by 2.
// ---------------------------------------------------------------------------
__global__ void k_bias(const float* __restrict__ Y,
                       float* __restrict__ qbias, float* __restrict__ kbias) {
    int tid = blockIdx.x * blockDim.x + threadIdx.x;           // 8*9*1024
    if (tid >= BB * ENT * NN) return;
    int pos = tid & (NN - 1);
    int be  = tid >> 10;
    int e = be % ENT, b = be / ENT;
    const float* yr = Y + ((size_t)b * NN + pos) * WPAD + 128;
    qbias[tid] = yr[2 * e]     * 0.5f;
    kbias[tid] = yr[2 * e + 1] * 0.5f;
}

// ---------------------------------------------------------------------------
// Kernel 4: QK GEMM tile + fused bias/mask/tril epilogue.
// One wave per (b, 16-row m-tile, 16-col n-tile). 2 WMMA (K=64), then 9
// entity planes written (302 MB total, written exactly once).
// ---------------------------------------------------------------------------
__global__ void k_logits(const __bf16* __restrict__ qwr, const __bf16* __restrict__ kwr,
                         const float* __restrict__ qbias, const float* __restrict__ kbias,
                         const float* __restrict__ mask, float* __restrict__ out) {
    int wave = (blockIdx.x * blockDim.x + threadIdx.x) >> 5;   // 0..32767
    int lane = threadIdx.x & 31;
    int b   = wave >> 12;
    int rem = wave & 4095;
    int mt = rem >> 6, nt = rem & 63;
    int lm = lane & 15, lh = lane >> 4;

    int m_a = mt * 16 + lm;                                    // A row (q position)
    int n_b = nt * 16 + lm;                                    // B col (k position)

    const __bf16* ap = qwr + ((size_t)b * NN + m_a) * INNER + lh * 16;
    const __bf16* bp = kwr + ((size_t)b * NN + n_b) * INNER + lh * 16;

    v8f acc = (v8f){0,0,0,0,0,0,0,0};
#pragma unroll
    for (int kt = 0; kt < 2; ++kt) {
        int kbase = kt * 32;
        v8bf alo = *(const v8bf*)(ap + kbase);
        v8bf ahi = *(const v8bf*)(ap + kbase + 8);
        v16bf a = __builtin_shufflevector(alo, ahi,
                     0,1,2,3,4,5,6,7,8,9,10,11,12,13,14,15);
        v16bf bm = *(const v16bf*)(bp + kbase);
        acc = __builtin_amdgcn_wmma_f32_16x16x32_bf16(
            false, a, false, bm, (short)0, acc, false, false);
    }

    // Epilogue. C layout: row = mrow0 + r, col = ncol.
    int ncol  = nt * 16 + lm;
    int mrow0 = mt * 16 + 8 * lh;
    float mcol = mask[b * NN + ncol];
    float mrow[8], base[8];
    {
        v4f t0 = *(const v4f*)(mask + b * NN + mrow0);
        v4f t1 = *(const v4f*)(mask + b * NN + mrow0 + 4);
#pragma unroll
        for (int r = 0; r < 4; ++r) { mrow[r] = t0[r]; mrow[r + 4] = t1[r]; }
    }
#pragma unroll
    for (int r = 0; r < 8; ++r) base[r] = acc[r] * 0.125f;     // 1/sqrt(64)

    size_t plane = (size_t)NN * NN;
    float* ob = out + (size_t)b * ENT * plane;
#pragma unroll
    for (int e = 0; e < ENT; ++e) {
        float kb = kbias[((size_t)b * ENT + e) * NN + ncol];
        const float* qbp = qbias + ((size_t)b * ENT + e) * NN + mrow0;
        float qb[8];
        {
            v4f t0 = *(const v4f*)(qbp);
            v4f t1 = *(const v4f*)(qbp + 4);
#pragma unroll
            for (int r = 0; r < 4; ++r) { qb[r] = t0[r]; qb[r + 4] = t1[r]; }
        }
        float* op = ob + (size_t)e * plane;
#pragma unroll
        for (int r = 0; r < 8; ++r) {
            int mrw = mrow0 + r;
            float v = base[r] + qb[r] + kb;
            v = v * mrow[r] - NEGV * (1.0f - mrow[r]);
            v = v * mcol    - NEGV * (1.0f - mcol);
            if (ncol < mrw) v -= NEGV;                          // tril(k=-1)
            op[(size_t)mrw * NN + ncol] = v;
        }
    }
}

// ---------------------------------------------------------------------------
extern "C" void kernel_launch(void* const* d_in, const int* in_sizes, int n_in,
                              void* d_out, int out_size, void* d_ws, size_t ws_size,
                              hipStream_t stream) {
    const float* lhs  = (const float*)d_in[0];   // (8,1024,768)
    const float* amsk = (const float*)d_in[1];   // (8,1024)
    const float* w1   = (const float*)d_in[2];   // (768,128)
    const float* b1   = (const float*)d_in[3];   // (128,)
    const float* w2   = (const float*)d_in[4];   // (768,18)
    const float* b2   = (const float*)d_in[5];   // (18,)
    float* out = (float*)d_out;                  // (8,9,1024,1024)

    // workspace carving (all sections naturally >=256B aligned)
    char* ws = (char*)d_ws;
    size_t offY  = 0;                                  // Y: 8192*160 f32
    size_t offWc = offY  + (size_t)BB*NN*WPAD*4;       // WcT: 160*768 bf16
    size_t offXb = offWc + (size_t)WPAD*HH*2;          // lhsb: 8192*768 bf16
    size_t offQ  = offXb + (size_t)BB*NN*HH*2;         // qw: 8192*64 bf16
    size_t offK  = offQ  + (size_t)BB*NN*INNER*2;      // kw: 8192*64 bf16
    size_t offQB = offK  + (size_t)BB*NN*INNER*2;      // qbias: 8*9*1024 f32
    size_t offKB = offQB + (size_t)BB*ENT*NN*4;        // kbias: 8*9*1024 f32

    float*  Y     = (float*)(ws + offY);
    __bf16* WcT   = (__bf16*)(ws + offWc);
    __bf16* lhsb  = (__bf16*)(ws + offXb);
    __bf16* qwr   = (__bf16*)(ws + offQ);
    __bf16* kwr   = (__bf16*)(ws + offK);
    float*  qbias = (float*)(ws + offQB);
    float*  kbias = (float*)(ws + offKB);

    const int thr = 256;
    // 0a: pack+transpose weights to bf16
    {
        int n = WPAD * HH;
        k_prep_w<<<(n + thr - 1) / thr, thr, 0, stream>>>(w1, w2, WcT);
    }
    // 0b: lhs -> bf16
    {
        int n = (BB * NN * HH) / 4;
        k_prep_x<<<(n + thr - 1) / thr, thr, 0, stream>>>(lhs, lhsb);
    }
    // 1: projection GEMM (512 waves, 8 waves/block)
    {
        int waves = (BB * NN) / 16;                    // 512
        k_proj<<<waves * 32 / thr, thr, 0, stream>>>(lhsb, WcT, b1, b2, Y);
    }
    // 2: RoPE split
    {
        int n = BB * NN * INNER;
        k_rope<<<(n + thr - 1) / thr, thr, 0, stream>>>(Y, qwr, kwr);
    }
    // 3: entity biases
    {
        int n = BB * ENT * NN;
        k_bias<<<(n + thr - 1) / thr, thr, 0, stream>>>(Y, qbias, kbias);
    }
    // 4: QK GEMM + fused epilogue (32768 waves)
    {
        int waves = BB * (NN / 16) * (NN / 16);        // 32768
        k_logits<<<waves * 32 / thr, thr, 0, stream>>>(qwr, kwr, qbias, kbias, amsk, out);
    }
}
